// DMCR_86466281603491
// MI455X (gfx1250) — matrix operand order repaired and checked by reading, not compiled, not run on hardware
//
#include <hip/hip_runtime.h>
#include <hip/hip_bf16.h>

// ---------------------------------------------------------------------------
// DMCR multi-criteria GCN forward for MI455X (gfx1250, wave32, WMMA).
//
// Pipeline per layer k (all buffers in (C,N,D) layout unless noted):
//   1) embBuf = 0;  SpMM scatter-add (atomics, L2-resident gather)
//   2) M_c = w_gcn @ diag(cri_c) @ W_gc[k]        (tiny, collapses 2 GEMMs)
//   3) hBuf = leaky(embBuf @ M_c)                  (WMMA bf16 hi/lo split)
//   4) attention per node (tanh-dominated, VALU):  preBuf, acc += pre
//   5) cri update
// acc lives in d_out in (n,c,d) order == users‖items flat layout.
// ---------------------------------------------------------------------------

#define N_USERS 50000
#define NNODE   100000
#define CC      5
#define DD      64
#define AA      32
#define LL      2
#define ALPHA   0.3f

typedef __attribute__((ext_vector_type(16))) __bf16 v16bf;
typedef __attribute__((ext_vector_type(8)))  float  v8f;

__device__ __forceinline__ float leaky(float x) { return x >= 0.f ? x : ALPHA * x; }

// ---------------------------------------------------------------------------
// Init: pre (C,N,D) and acc (n,c,d in d_out) from user/item embeddings; cri.
// ---------------------------------------------------------------------------
__global__ void init_kernel(const float* __restrict__ ue, const float* __restrict__ ie,
                            const float* __restrict__ cri_emb,
                            float* __restrict__ pre, float* __restrict__ acc,
                            float* __restrict__ criCur, float* __restrict__ criSum)
{
    const size_t total  = (size_t)CC * NNODE * DD;
    const size_t stride = (size_t)gridDim.x * blockDim.x;
    const size_t gid    = (size_t)blockIdx.x * blockDim.x + threadIdx.x;
    if (gid < CC * DD) { float v = cri_emb[gid]; criCur[gid] = v; criSum[gid] = v; }
    for (size_t idx = gid; idx < total; idx += stride) {
        int    c   = (int)(idx / ((size_t)NNODE * DD));
        size_t rem = idx - (size_t)c * NNODE * DD;
        int    n   = (int)(rem / DD);
        int    d   = (int)(rem % DD);
        float  v   = (n < N_USERS) ? ue[((size_t)n * CC + c) * DD + d]
                                   : ie[((size_t)(n - N_USERS) * CC + c) * DD + d];
        pre[idx] = v;
        acc[((size_t)n * CC + c) * DD + d] = v;
    }
}

// ---------------------------------------------------------------------------
// SpMM: Y[c, rows[e], :] += vals[e] * X[c, cols[e], :].  One wave per edge,
// lane handles 2 dwords of D=64.  Prefetch next gather row (L2-resident).
// Grid: (blocks, CC).
// ---------------------------------------------------------------------------
__global__ void spmm_kernel(const int* __restrict__ rows, const int* __restrict__ cols,
                            const float* __restrict__ vals, const float* __restrict__ X,
                            float* __restrict__ Y, int Eper)
{
    const int c      = blockIdx.y;
    const int lane   = threadIdx.x & 31;
    const int d0     = lane * 2;
    int gwave  = (int)((blockIdx.x * blockDim.x + threadIdx.x) >> 5);
    int nwaves = (int)((gridDim.x * blockDim.x) >> 5);

    const int*   rC = rows + (size_t)c * Eper;
    const int*   cCo= cols + (size_t)c * Eper;
    const float* vC = vals + (size_t)c * Eper;
    const float* Xc = X + (size_t)c * NNODE * DD;
    float*       Yc = Y + (size_t)c * NNODE * DD;

    for (int e = gwave; e < Eper; e += nwaves) {
        int   r = rC[e];
        int   s = cCo[e];
        float v = vC[e];
        int e2 = e + nwaves;
        if (e2 < Eper) {
            int s2 = cCo[e2];
            __builtin_prefetch(Xc + (size_t)s2 * DD + d0, 0, 1);  // global_prefetch_b8
        }
        const float2 xv = *(const float2*)(Xc + (size_t)s * DD + d0);
        float* yp = Yc + (size_t)r * DD + d0;
        unsafeAtomicAdd(yp,     v * xv.x);   // global_atomic_add_f32
        unsafeAtomicAdd(yp + 1, v * xv.y);
    }
}

// ---------------------------------------------------------------------------
// M_c[p][q] = sum_d w_gcn[p][d] * cri[c][d] * Wg[d][q]   (C x 64 x 64)
// ---------------------------------------------------------------------------
__global__ void build_M_kernel(const float* __restrict__ w_gcn, const float* __restrict__ Wg,
                               const float* __restrict__ cri, float* __restrict__ M)
{
    int c  = blockIdx.y;
    int pq = blockIdx.x * blockDim.x + threadIdx.x;
    if (pq >= DD * DD) return;
    int p = pq >> 6, q = pq & 63;
    float s = 0.f;
    for (int d = 0; d < DD; ++d)
        s = fmaf(w_gcn[p * DD + d] * cri[c * DD + d], Wg[d * DD + q], s);
    M[((size_t)c * DD + p) * DD + q] = s;
}

// ---------------------------------------------------------------------------
// H[c] = leaky(X[c] @ M_c), (N x 64)(64 x 64) per criterion.
// Wave computes a 16x64 output tile via v_wmma_f32_16x16x32_bf16.
// Split-bf16 (hi + lo residual) restores ~fp32 accuracy: D += Ah*Bh + Ah*Bl + Al*Bh.
// Block = 128 threads (4 waves) -> 64-row tile.  Grid: (ceil(N/64), CC).
// ---------------------------------------------------------------------------
__global__ void __launch_bounds__(128)
gemm_leaky_wmma(const float* __restrict__ X, const float* __restrict__ M,
                float* __restrict__ H)
{
    const int c       = blockIdx.y;
    const int rowTile = blockIdx.x * 64;
    const int tid     = threadIdx.x;
    const int wave    = tid >> 5;
    const int lane    = tid & 31;

    // Bt[n*64+k] = M_c[k][n] as bf16 hi + residual lo (transposed for 32B frag loads)
    __shared__ __align__(32) __bf16 Bt_hi[DD * DD];
    __shared__ __align__(32) __bf16 Bt_lo[DD * DD];
    const float* Mc = M + (size_t)c * DD * DD;
    for (int idx = tid; idx < DD * DD; idx += 128) {
        int n = idx >> 6, k = idx & 63;
        float mv = Mc[k * DD + n];
        __bf16 hv = (__bf16)mv;
        Bt_hi[idx] = hv;
        Bt_lo[idx] = (__bf16)(mv - (float)hv);
    }
    __syncthreads();

    const int  m     = lane & 15;
    const int  hi    = lane >> 4;                  // half-wave select
    const int  row   = rowTile + wave * 16 + m;
    const bool rowOK = row < NNODE;
    const float* Xrow = X + ((size_t)c * NNODE + row) * DD;

    v8f acc0 = {}, acc1 = {}, acc2 = {}, acc3 = {};

    #pragma unroll
    for (int kc = 0; kc < 2; ++kc) {
        // A fragment (contiguous k order, same as B): lane row = l%16,
        // K = kc*32 + 16*(l>=16) + {0..15}
        v16bf a_hi, a_lo;
        #pragma unroll
        for (int e = 0; e < 16; ++e) { a_hi[e] = (__bf16)0.0f; a_lo[e] = (__bf16)0.0f; }
        if (rowOK) {
            const int kb = kc * 32 + hi * 16;
            float4 f0 = *(const float4*)(Xrow + kb);
            float4 f1 = *(const float4*)(Xrow + kb + 4);
            float4 f2 = *(const float4*)(Xrow + kb + 8);
            float4 f3 = *(const float4*)(Xrow + kb + 12);
            float xv[16] = { f0.x,f0.y,f0.z,f0.w, f1.x,f1.y,f1.z,f1.w,
                             f2.x,f2.y,f2.z,f2.w, f3.x,f3.y,f3.z,f3.w };
            #pragma unroll
            for (int e = 0; e < 16; ++e) {
                __bf16 hv = (__bf16)xv[e];
                a_hi[e] = hv;
                a_lo[e] = (__bf16)(xv[e] - (float)hv);
            }
        }
        // B fragments: lane col = l%16, K = kc*32 + 16*(l>=16) + e  (contiguous in Bt)
        const int kbB = kc * 32 + hi * 16;
        #pragma unroll
        for (int nc = 0; nc < 4; ++nc) {
            v16bf bh = *(const v16bf*)(&Bt_hi[(nc * 16 + m) * DD + kbB]);
            v16bf bl = *(const v16bf*)(&Bt_lo[(nc * 16 + m) * DD + kbB]);
            v8f* accp = (nc == 0) ? &acc0 : (nc == 1) ? &acc1 : (nc == 2) ? &acc2 : &acc3;
            v8f acc = *accp;
            acc = __builtin_amdgcn_wmma_f32_16x16x32_bf16(false, a_hi, false, bh, (short)0, acc, false, false);
            acc = __builtin_amdgcn_wmma_f32_16x16x32_bf16(false, a_hi, false, bl, (short)0, acc, false, false);
            acc = __builtin_amdgcn_wmma_f32_16x16x32_bf16(false, a_lo, false, bh, (short)0, acc, false, false);
            *accp = acc;
        }
    }

    // C/D layout: VGPR r -> row r + 8*(l>=16), col = l%16
    float* Hc = H + (size_t)c * NNODE * DD;
    #pragma unroll
    for (int r = 0; r < 8; ++r) {
        int orow = rowTile + wave * 16 + hi * 8 + r;
        if (orow < NNODE) {
            float* p = Hc + (size_t)orow * DD + m;
            p[0]  = leaky(acc0[r]);
            p[16] = leaky(acc1[r]);
            p[32] = leaky(acc2[r]);
            p[48] = leaky(acc3[r]);
        }
    }
}

// ---------------------------------------------------------------------------
// Attention (tanh-bound, fp32 VALU).  One wave per node, lane = attention unit a.
//   s[i][c] = sum_a tanh(sum_d h[c][d] W1[i][d][a]) W2[i][a]
//   attn    = softmax_c(s);  pre[i][d] = leaky(sum_c attn[i][c] h[c][d])
// Writes Pre (C,N,D) and accumulates Acc (n,c,d) = d_out.
// ---------------------------------------------------------------------------
__global__ void __launch_bounds__(256)
attention_kernel(const float* __restrict__ H, const float* __restrict__ W1,
                 const float* __restrict__ W2, float* __restrict__ Pre,
                 float* __restrict__ Acc)
{
    __shared__ float sW1[CC * DD * AA];      // 40 KB, [(i*64+d)*32 + a]
    __shared__ float sW2[CC * AA];
    __shared__ float sH[8 * CC * DD];        // per-wave node embedding

    const int tid  = threadIdx.x;
    const int lane = tid & 31;
    const int wv   = tid >> 5;

    for (int t = tid; t < CC * DD * AA; t += blockDim.x) sW1[t] = W1[t];
    for (int t = tid; t < CC * AA;      t += blockDim.x) sW2[t] = W2[t];
    __syncthreads();

    const int totalWaves = (int)((gridDim.x * blockDim.x) >> 5);
    const int gwave      = (int)((blockIdx.x * blockDim.x + tid) >> 5);
    const int iters      = (NNODE + totalWaves - 1) / totalWaves;
    float* h = sH + wv * (CC * DD);

    for (int it = 0; it < iters; ++it) {
        const int  n  = it * totalWaves + gwave;
        const bool ok = (n < NNODE);                  // wave-uniform
        __syncthreads();                              // guard sH reuse
        if (ok) {
            for (int t = lane; t < CC * DD; t += 32)
                h[t] = H[((size_t)(t >> 6) * NNODE + n) * DD + (t & 63)];
        }
        __syncthreads();
        if (ok) {
            float s[CC][CC];
            #pragma unroll
            for (int i = 0; i < CC; ++i)
                #pragma unroll
                for (int c2 = 0; c2 < CC; ++c2) s[i][c2] = 0.f;

            #pragma unroll 4
            for (int d = 0; d < DD; ++d) {
                float w[CC], hh[CC];
                #pragma unroll
                for (int i = 0; i < CC; ++i)  w[i]  = sW1[(i * DD + d) * AA + lane];
                #pragma unroll
                for (int c2 = 0; c2 < CC; ++c2) hh[c2] = h[c2 * DD + d];
                #pragma unroll
                for (int i = 0; i < CC; ++i)
                    #pragma unroll
                    for (int c2 = 0; c2 < CC; ++c2)
                        s[i][c2] = fmaf(hh[c2], w[i], s[i][c2]);
            }

            // tanh * W2, cross-lane reduce over a (wave32 butterfly)
            #pragma unroll
            for (int i = 0; i < CC; ++i) {
                float w2 = sW2[i * AA + lane];
                #pragma unroll
                for (int c2 = 0; c2 < CC; ++c2) {
                    float t = tanhf(s[i][c2]) * w2;
                    #pragma unroll
                    for (int off = 16; off > 0; off >>= 1) t += __shfl_xor(t, off, 32);
                    s[i][c2] = t;
                }
            }
            // softmax over c (5 values, replicated in every lane)
            #pragma unroll
            for (int i = 0; i < CC; ++i) {
                float mx = s[i][0];
                #pragma unroll
                for (int c2 = 1; c2 < CC; ++c2) mx = fmaxf(mx, s[i][c2]);
                float sum = 0.f;
                #pragma unroll
                for (int c2 = 0; c2 < CC; ++c2) { s[i][c2] = __expf(s[i][c2] - mx); sum += s[i][c2]; }
                float inv = 1.f / sum;
                #pragma unroll
                for (int c2 = 0; c2 < CC; ++c2) s[i][c2] *= inv;
            }
            // combine, leaky, write pre and accumulate into acc (d_out)
            #pragma unroll
            for (int half = 0; half < 2; ++half) {
                int d = lane + half * 32;
                float hv[CC];
                #pragma unroll
                for (int c2 = 0; c2 < CC; ++c2) hv[c2] = h[c2 * DD + d];
                #pragma unroll
                for (int i = 0; i < CC; ++i) {
                    float v = 0.f;
                    #pragma unroll
                    for (int c2 = 0; c2 < CC; ++c2) v = fmaf(s[i][c2], hv[c2], v);
                    v = leaky(v);
                    Pre[((size_t)i * NNODE + n) * DD + d] = v;
                    size_t oi = ((size_t)n * CC + i) * DD + d;
                    Acc[oi] += v;
                }
            }
        }
    }
}

// ---------------------------------------------------------------------------
// cri_next = leaky(cri_cur @ W_rel[k]);  cri_sum += cri_next
// ---------------------------------------------------------------------------
__global__ void cri_update_kernel(const float* __restrict__ criIn, const float* __restrict__ Wrel,
                                  float* __restrict__ criOut, float* __restrict__ criSum)
{
    int idx = blockIdx.x * blockDim.x + threadIdx.x;
    if (idx >= CC * DD) return;
    int c = idx >> 6, q = idx & 63;
    float s = 0.f;
    for (int d = 0; d < DD; ++d) s = fmaf(criIn[c * DD + d], Wrel[d * DD + q], s);
    s = leaky(s);
    criOut[idx] = s;
    criSum[idx] += s;
}

// ---------------------------------------------------------------------------
// Finalize: scale acc (already in out, (n,c,d)) by 1/3, zero pad row, cri_mean.
// ---------------------------------------------------------------------------
__global__ void finalize_kernel(float* __restrict__ out, const float* __restrict__ criSum)
{
    const size_t accElems = (size_t)NNODE * CC * DD;   // 32,000,000
    const size_t padElems = CC * DD;                   // items zero row
    const size_t total    = accElems + padElems + CC * DD;
    const float  inv3     = 1.f / 3.f;
    const size_t stride   = (size_t)gridDim.x * blockDim.x;
    for (size_t idx = (size_t)blockIdx.x * blockDim.x + threadIdx.x; idx < total; idx += stride) {
        if (idx < accElems)                  out[idx] *= inv3;
        else if (idx < accElems + padElems)  out[idx]  = 0.f;
        else                                 out[idx]  = criSum[idx - accElems - padElems] * inv3;
    }
}

// ---------------------------------------------------------------------------
extern "C" void kernel_launch(void* const* d_in, const int* in_sizes, int n_in,
                              void* d_out, int out_size, void* d_ws, size_t ws_size,
                              hipStream_t stream)
{
    const int*   rows     = (const int*)d_in[0];
    const int*   cols     = (const int*)d_in[1];
    const float* vals     = (const float*)d_in[2];
    const float* user_emb = (const float*)d_in[3];
    const float* item_emb = (const float*)d_in[4];
    const float* cri_emb  = (const float*)d_in[5];
    const float* w_gcn    = (const float*)d_in[6];
    const float* W_gc     = (const float*)d_in[7];
    const float* W_rel    = (const float*)d_in[8];
    const float* W1       = (const float*)d_in[9];
    const float* W2       = (const float*)d_in[10];

    const int Eper = in_sizes[0] / CC;

    float*       acc = (float*)d_out;            // acc lives in the output buffer
    float*       ws  = (float*)d_ws;
    const size_t big = (size_t)CC * NNODE * DD;  // 32,000,000 floats (128 MB)
    float* preBuf = ws;
    float* embBuf = preBuf + big;
    float* hBuf   = embBuf + big;
    float* Mbuf   = hBuf + big;                  // C*64*64
    float* criA   = Mbuf + (size_t)CC * DD * DD;
    float* criB   = criA + CC * DD;
    float* criSum = criB + CC * DD;

    init_kernel<<<2048, 256, 0, stream>>>(user_emb, item_emb, cri_emb,
                                          preBuf, acc, criA, criSum);

    float* criCur  = criA;
    float* criNext = criB;
    for (int k = 0; k < LL; ++k) {
        hipMemsetAsync(embBuf, 0, big * sizeof(float), stream);
        spmm_kernel<<<dim3(4096, CC), 256, 0, stream>>>(rows, cols, vals, preBuf, embBuf, Eper);
        build_M_kernel<<<dim3(16, CC), 256, 0, stream>>>(w_gcn, W_gc + (size_t)k * DD * DD,
                                                         criCur, Mbuf);
        gemm_leaky_wmma<<<dim3((NNODE + 63) / 64, CC), 128, 0, stream>>>(embBuf, Mbuf, hBuf);
        attention_kernel<<<1024, 256, 0, stream>>>(hBuf, W1, W2, preBuf, acc);
        cri_update_kernel<<<2, 256, 0, stream>>>(criCur, W_rel + (size_t)k * DD * DD,
                                                 criNext, criSum);
        float* t = criCur; criCur = criNext; criNext = t;
    }

    finalize_kernel<<<4096, 256, 0, stream>>>(acc, criSum);
}
